// XANES_E3GNN_2293512536748
// MI455X (gfx1250) — compile-verified, matrix operand order checked by reading, not run on hardware
//
#include <hip/hip_runtime.h>

// XANES E3GNN forward for gfx1250 (MI455X).
// Dense channel-mixing GEMMs use v_wmma_f32_16x16x32_bf16 (one 16-row tile / wave32),
// with compile-time K so A fragments are straight-line b128 loads + cvt_pk_bf16, and
// weights pre-packed into fragment-ordered bf16 (single 32B vector load per B frag).
// Edge scatter uses L2 atomics (accumulators fit the 192MB L2).
// Per-edge radial-MLP weight expansion is fused (LDS-staged rW2 + wave shuffles).

typedef __attribute__((ext_vector_type(16))) __bf16 v16bf;
typedef __attribute__((ext_vector_type(8)))  float  v8f;

__device__ __forceinline__ float silu_f(float x) { return x / (1.0f + __expf(-x)); }

__device__ __forceinline__ unsigned f2ord(float f) {
  unsigned u = __float_as_uint(f);
  return (u & 0x80000000u) ? ~u : (u | 0x80000000u);
}
__device__ __forceinline__ float ord2f(unsigned o) {
  unsigned u = (o & 0x80000000u) ? (o & 0x7fffffffu) : ~o;
  return __uint_as_float(u);
}
static inline unsigned h_f2ord(float f) {
  unsigned u; __builtin_memcpy(&u, &f, 4);
  return (u & 0x80000000u) ? ~u : (u | 0x80000000u);
}

// ---------------- WMMA fragment loaders (ISA 7.12.2 16-bit layouts) ----------------
// A (16x32 MxK bf16): lane holds row lane&15; K elements are two contiguous 8-runs:
// [kbase+khi*8, +8) and [kbase+16+khi*8, +8).  K is compile-time and a multiple of 16,
// so each 8-run is entirely valid or entirely zero -> branch-free b128 loads.
template <int K>
__device__ __forceinline__ v16bf load_a_frag(const float* __restrict__ A, int lda,
                                             int row, int kbase, int khi) {
  v16bf a;
#pragma unroll
  for (int c = 0; c < 2; ++c) {
    if (kbase + c * 16 < K) {
      const float* p = A + (size_t)row * lda + (kbase + c * 16 + khi * 8);
      float4 f0 = *(const float4*)(p);
      float4 f1 = *(const float4*)(p + 4);
      a[c * 8 + 0] = (__bf16)f0.x;  a[c * 8 + 1] = (__bf16)f0.y;
      a[c * 8 + 2] = (__bf16)f0.z;  a[c * 8 + 3] = (__bf16)f0.w;
      a[c * 8 + 4] = (__bf16)f1.x;  a[c * 8 + 5] = (__bf16)f1.y;
      a[c * 8 + 6] = (__bf16)f1.z;  a[c * 8 + 7] = (__bf16)f1.w;
    } else {
#pragma unroll
      for (int i = 0; i < 8; ++i) a[c * 8 + i] = (__bf16)0.0f;
    }
  }
  return a;
}

// Pack W[K,NC] (f32 row-major) into fragment-ordered bf16:
// P[((kt*numNt + nt)*32 + lane)*16 + e] = bf16 of B-matrix element for that lane/frag.
__global__ void pack_w_kernel(const float* __restrict__ W, __bf16* __restrict__ P,
                              int K, int NCc) {
  int numNt = NCc >> 4;
  int numKt = (K + 31) >> 5;
  int total = numKt * numNt * 32;
  int tid = blockIdx.x * blockDim.x + threadIdx.x;
  if (tid >= total) return;
  int lane = tid & 31;
  int tile = tid >> 5;
  int nt = tile % numNt, kt = tile / numNt;
  int khi = lane >> 4;
  int col = nt * 16 + (lane & 15);
  v16bf vals;
#pragma unroll
  for (int p = 0; p < 8; ++p) {
    int kk = kt * 32 + (p & 3) * 2 + ((p >= 4) ? 16 : 0) + khi * 8;
    float x0 = (kk < K) ? W[(size_t)kk * NCc + col] : 0.0f;
    float x1 = (kk + 1 < K) ? W[(size_t)(kk + 1) * NCc + col] : 0.0f;
    vals[2 * p]     = (__bf16)x0;
    vals[2 * p + 1] = (__bf16)x1;
  }
  *(v16bf*)(P + (size_t)tid * 16) = vals;
}

// OUT[M,NC] = act( A[M,KA]@WA (+ B[M,KB]@WB) (+ bias) ), one 16-row tile per wave32.
// WA/WB are fragment-ordered bf16 (from pack_w_kernel).
template <int KA, int KB, int NC, bool SILU, bool HASB, bool HASBIAS, bool GATHERA>
__global__ __launch_bounds__(128) void gemm16_kernel(
    const float* __restrict__ A, int lda, const __bf16* __restrict__ WA,
    const float* __restrict__ B, int ldb, const __bf16* __restrict__ WB,
    const float* __restrict__ bias, const int* __restrict__ rowidx,
    float* __restrict__ OUT, int ldo, int M) {
  constexpr int NT  = NC >> 4;
  constexpr int KTA = (KA + 31) >> 5;
  constexpr int KTB = HASB ? ((KB + 31) >> 5) : 1;

  int tiles = (M + 15) >> 4;
  int wid = blockIdx.x * 4 + (threadIdx.x >> 5);
  if (wid >= tiles) return;
  int lane = threadIdx.x & 31;
  int khi = lane >> 4, l16 = lane & 15;
  int row0 = wid << 4;
  int row = row0 + l16; if (row >= M) row = M - 1;
  int arow = GATHERA ? rowidx[row] : row;

  v16bf afA[KTA];
#pragma unroll
  for (int kt = 0; kt < KTA; ++kt) afA[kt] = load_a_frag<KA>(A, lda, arow, kt * 32, khi);
  v16bf afB[KTB];
  if constexpr (HASB) {
#pragma unroll
    for (int kt = 0; kt < KTB; ++kt) afB[kt] = load_a_frag<KB>(B, ldb, arow, kt * 32, khi);
  }

#pragma unroll
  for (int nt = 0; nt < NT; ++nt) {
    v8f acc = {0.f, 0.f, 0.f, 0.f, 0.f, 0.f, 0.f, 0.f};
#pragma unroll
    for (int kt = 0; kt < KTA; ++kt) {
      v16bf bf = *(const v16bf*)(WA + ((size_t)(kt * NT + nt) * 32 + lane) * 16);
      acc = __builtin_amdgcn_wmma_f32_16x16x32_bf16(false, afA[kt], false, bf,
                                                    (short)0, acc, false, false);
    }
    if constexpr (HASB) {
#pragma unroll
      for (int kt = 0; kt < KTB; ++kt) {
        v16bf bf = *(const v16bf*)(WB + ((size_t)(kt * NT + nt) * 32 + lane) * 16);
        acc = __builtin_amdgcn_wmma_f32_16x16x32_bf16(false, afB[kt], false, bf,
                                                      (short)0, acc, false, false);
      }
    }
    int ocol = (nt << 4) + l16;
    float bc = HASBIAS ? bias[ocol] : 0.0f;
#pragma unroll
    for (int r = 0; r < 8; ++r) {
      int orow = row0 + r + khi * 8;   // C/D layout: lanes>=16 hold rows 8..15
      if (orow < M) {
        float x = acc[r] + bc;
        if (SILU) x = silu_f(x);
        OUT[(size_t)orow * ldo + ocol] = x;
      }
    }
  }
}

// ---------------- geometry: spherical harmonics + radial basis ----------------
__global__ void edge_geom_kernel(const float* __restrict__ pos, const int* __restrict__ ei,
                                 float* __restrict__ Y, float* __restrict__ rb, int E) {
  int e = blockIdx.x * blockDim.x + threadIdx.x;
  if (e >= E) return;
  int s = ei[e], d = ei[E + e];
  float vx = pos[d * 3 + 0] - pos[s * 3 + 0];
  float vy = pos[d * 3 + 1] - pos[s * 3 + 1];
  float vz = pos[d * 3 + 2] - pos[s * 3 + 2];
  float len = sqrtf(vx * vx + vy * vy + vz * vz);
  float inv = 1.0f / (len + 1e-9f);
  float x = vx * inv, y = vy * inv, z = vz * inv;
  const float s3 = 1.7320508075688772f, s5 = 2.23606797749979f, s15 = 3.872983346207417f;
  float* yp = Y + (size_t)e * 9;
  yp[0] = 1.0f;
  yp[1] = s3 * x;  yp[2] = s3 * y;  yp[3] = s3 * z;
  yp[4] = s15 * x * y;  yp[5] = s15 * y * z;
  yp[6] = 0.5f * s5 * (3.0f * z * z - 1.0f);
  yp[7] = s15 * x * z;
  yp[8] = 0.5f * s15 * (x * x - y * y);
  const float sigma = 5.0f / 9.0f;
#pragma unroll
  for (int r = 0; r < 10; ++r) {
    float t = (len - (float)r * sigma) / sigma;
    rb[(size_t)e * 10 + r] = __expf(-0.5f * t * t);
  }
}

__global__ void hs_init_kernel(const int* __restrict__ z, const float* __restrict__ embed,
                               float* __restrict__ hs, int N) {
  int i = blockIdx.x * blockDim.x + threadIdx.x;
  if (i >= N * 64) return;
  int n = i >> 6, c = i & 63;
  hs[i] = embed[(size_t)z[n] * 64 + c];
}

__global__ void fill_f32(float* __restrict__ p, float v, size_t n) {
  size_t i = (size_t)blockIdx.x * blockDim.x + threadIdx.x;
  size_t st = (size_t)gridDim.x * blockDim.x;
  for (; i < n; i += st) p[i] = v;
}
__global__ void fill_u32(unsigned* __restrict__ p, unsigned v, size_t n) {
  size_t i = (size_t)blockIdx.x * blockDim.x + threadIdx.x;
  size_t st = (size_t)gridDim.x * blockDim.x;
  for (; i < n; i += st) p[i] = v;
}

// ---------------- fused edge-weight + message + scatter (one wave32 per edge) ----------------
__global__ __launch_bounds__(256) void message_kernel(
    const int* __restrict__ ei, const float* __restrict__ Y, const float* __restrict__ rbuf,
    const float* __restrict__ hs, const float* __restrict__ hv, const float* __restrict__ ht,
    const float* __restrict__ rW1, const float* __restrict__ rb1, const float* __restrict__ rW2,
    float* __restrict__ a0, float* __restrict__ a1, float* __restrict__ a2, int E) {
  __shared__ float s_rW2[32 * 320];
  __shared__ float s_rW1[10 * 32];
  __shared__ float s_rb1[32];
  for (int i = threadIdx.x; i < 32 * 320; i += 256) s_rW2[i] = rW2[i];
  for (int i = threadIdx.x; i < 10 * 32; i += 256) s_rW1[i] = rW1[i];
  if (threadIdx.x < 32) s_rb1[threadIdx.x] = rb1[threadIdx.x];
  __syncthreads();

  const int lane = threadIdx.x & 31;
  const int e = blockIdx.x * 8 + (threadIdx.x >> 5);
  const bool valid = (e < E);
  const int ec = valid ? e : 0;
  const int src = ei[ec];
  const int dst = ei[E + ec];

  const float y0 = Y[(size_t)ec * 9 + 0];
  const float y1x = Y[(size_t)ec * 9 + 1], y1y = Y[(size_t)ec * 9 + 2], y1z = Y[(size_t)ec * 9 + 3];
  float y2[5];
#pragma unroll
  for (int i = 0; i < 5; ++i) y2[i] = Y[(size_t)ec * 9 + 4 + i];

  // hidden = silu(rb @ rW1 + rb1), one channel per lane
  float acc = s_rb1[lane];
#pragma unroll
  for (int r = 0; r < 10; ++r) acc += rbuf[(size_t)ec * 10 + r] * s_rW1[r * 32 + lane];
  const float hidden = silu_f(acc);

  // lazy weight expansion: each lane computes only the 11 rW2 columns it consumes
  const int lt = lane & 15;
  float w0a = 0.f, w0b = 0.f, w0v = 0.f, w0t = 0.f;
  float w1sa = 0.f, w1sb = 0.f, w1v = 0.f, w1x = 0.f;
  float w2sa = 0.f, w2sb = 0.f, w2t = 0.f;
  for (int j = 0; j < 32; ++j) {
    const float hj = __shfl(hidden, j, 32);
    const float* r2 = &s_rW2[j * 320];
    w0a  += hj * r2[lane];        w0b  += hj * r2[32 + lane];
    w0v  += hj * r2[64 + lane];   w0t  += hj * r2[96 + lt];
    w1sa += hj * r2[112 + lane];  w1sb += hj * r2[144 + lane];
    w1v  += hj * r2[176 + lane];  w1x  += hj * r2[208 + lane];
    w2sa += hj * r2[240 + lane];  w2sb += hj * r2[272 + lane];
    w2t  += hj * r2[304 + lt];
  }

  // gather source features (channel-per-lane, coalesced)
  const float s_lo = hs[(size_t)src * 64 + lane];
  const float s_hi = hs[(size_t)src * 64 + 32 + lane];
  const float vx = hv[((size_t)src * 3 + 0) * 32 + lane];
  const float vy = hv[((size_t)src * 3 + 1) * 32 + lane];
  const float vz = hv[((size_t)src * 3 + 2) * 32 + lane];
  float te[5];
#pragma unroll
  for (int d = 0; d < 5; ++d) te[d] = ht[((size_t)src * 5 + d) * 16 + lt];

  if (valid) {
    // m0 -> a0[dst, 112]
    float* p0 = a0 + (size_t)dst * 112;
    atomicAdd(p0 + lane,      w0a * s_lo * y0);
    atomicAdd(p0 + 32 + lane, w0b * s_hi * y0);
    const float vdot = vx * y1x + vy * y1y + vz * y1z;
    atomicAdd(p0 + 64 + lane, w0v * vdot);
    if (lane < 16) {
      const float tdot = te[0]*y2[0] + te[1]*y2[1] + te[2]*y2[2] + te[3]*y2[3] + te[4]*y2[4];
      atomicAdd(p0 + 96 + lane, w0t * tdot);
    }
    // m1 -> a1[(dst*3+d), 128]   (cross = v_e x y1)
    const float cx = vy * y1z - vz * y1y;
    const float cy = vz * y1x - vx * y1z;
    const float cz = vx * y1y - vy * y1x;
    const float yv[3] = {y1x, y1y, y1z};
    const float vv[3] = {vx, vy, vz};
    const float cc[3] = {cx, cy, cz};
#pragma unroll
    for (int d = 0; d < 3; ++d) {
      float* p1 = a1 + ((size_t)dst * 3 + d) * 128;
      atomicAdd(p1 + lane,      w1sa * s_lo * yv[d]);
      atomicAdd(p1 + 32 + lane, w1sb * s_hi * yv[d]);
      atomicAdd(p1 + 64 + lane, w1v * vv[d] * y0);
      atomicAdd(p1 + 96 + lane, w1x * cc[d]);
    }
    // m2 -> a2[(dst*5+d), 80]
#pragma unroll
    for (int d = 0; d < 5; ++d) {
      float* p2 = a2 + ((size_t)dst * 5 + d) * 80;
      atomicAdd(p2 + lane,      w2sa * s_lo * y2[d]);
      atomicAdd(p2 + 32 + lane, w2sb * s_hi * y2[d]);
      if (lane < 16) atomicAdd(p2 + 64 + lane, w2t * te[d] * y0);
    }
  }
}

// ---------------- readout ----------------
__global__ void logit_kernel(const int* __restrict__ batch, const float* __restrict__ q,
                             const float* __restrict__ k, float* __restrict__ logit,
                             unsigned* __restrict__ gmax, int N) {
  int i = blockIdx.x * blockDim.x + threadIdx.x;
  if (i >= N) return;
  int g = batch[i];
  const float* qp = q + (size_t)g * 64;
  const float* kp = k + (size_t)i * 64;
  float s = 0.f;
#pragma unroll 8
  for (int c = 0; c < 64; ++c) s += qp[c] * kp[c];
  s *= 0.125f;  // 1/sqrt(64)
  logit[i] = s;
  atomicMax(gmax + g, f2ord(s));
}

__global__ void expsum_kernel(const int* __restrict__ batch, const float* __restrict__ logit,
                              const unsigned* __restrict__ gmax, const float* __restrict__ v,
                              float* __restrict__ gden, float* __restrict__ gc, int N) {
  int i = blockIdx.x * blockDim.x + threadIdx.x;
  if (i >= N) return;
  int g = batch[i];
  float ex = __expf(logit[i] - ord2f(gmax[g]));
  atomicAdd(gden + g, ex);
  const float* vp = v + (size_t)i * 64;
  float* cp = gc + (size_t)g * 64;
#pragma unroll 4
  for (int c = 0; c < 64; ++c) atomicAdd(cp + c, ex * vp[c]);
}

__global__ void zr_kernel(const int* __restrict__ aidx, const float* __restrict__ hs,
                          const float* __restrict__ hv, const float* __restrict__ ht,
                          const float* __restrict__ gc, const float* __restrict__ gden,
                          float* __restrict__ zr, int G) {
  int idx = blockIdx.x * blockDim.x + threadIdx.x;
  if (idx >= G * 176) return;
  int g = idx / 176, j = idx - g * 176;
  int a = aidx[g];
  float val;
  if (j < 64) {
    val = hs[(size_t)a * 64 + j];
  } else if (j < 128) {
    val = gc[(size_t)g * 64 + (j - 64)] / fmaxf(gden[g], 1e-9f);
  } else if (j < 160) {
    int c = j - 128; float s = 0.f;
#pragma unroll
    for (int d = 0; d < 3; ++d) { float t = hv[((size_t)a * 3 + d) * 32 + c]; s += t * t; }
    val = s;
  } else {
    int c = j - 160; float s = 0.f;
#pragma unroll
    for (int d = 0; d < 5; ++d) { float t = ht[((size_t)a * 5 + d) * 16 + c]; s += t * t; }
    val = s;
  }
  zr[idx] = val;
}

// ---------------- host launcher ----------------
extern "C" void kernel_launch(void* const* d_in, const int* in_sizes, int n_in,
                              void* d_out, int out_size, void* d_ws, size_t ws_size,
                              hipStream_t stream) {
  (void)n_in; (void)out_size; (void)ws_size;
  const int*   z     = (const int*)d_in[0];
  const float* pos   = (const float*)d_in[1];
  const int*   ei    = (const int*)d_in[2];
  const int*   batch = (const int*)d_in[3];
  const int*   aidx  = (const int*)d_in[4];
  const float* embed = (const float*)d_in[5];
  const float* rW1   = (const float*)d_in[6];
  const float* rb1   = (const float*)d_in[7];
  const float* rW2   = (const float*)d_in[8];
  const float* Ws    = (const float*)d_in[9];
  const float* Wss   = (const float*)d_in[10];
  const float* Wv    = (const float*)d_in[11];
  const float* Wvv   = (const float*)d_in[12];
  const float* Wt    = (const float*)d_in[13];
  const float* Wtt   = (const float*)d_in[14];
  const float* Wq    = (const float*)d_in[15];
  const float* Wk    = (const float*)d_in[16];
  const float* Wvp   = (const float*)d_in[17];
  const float* Wr1   = (const float*)d_in[18];
  const float* br1   = (const float*)d_in[19];
  const float* Wr2   = (const float*)d_in[20];
  const float* br2   = (const float*)d_in[21];

  const int N = in_sizes[0];
  const int E = in_sizes[2] / 2;
  const int G = in_sizes[4];

  float* ws = (float*)d_ws;
  size_t off = 0;
  auto alloc = [&](size_t n) {            // 32B-aligned float allocations
    n = (n + 7) & ~(size_t)7;
    float* p = ws + off; off += n; return p;
  };
  float* Ybuf = alloc((size_t)E * 9);
  float* rbuf = alloc((size_t)E * 10);
  float* hsbuf[2] = { alloc((size_t)N * 64), alloc((size_t)N * 64) };
  float* hvbuf[2] = { alloc((size_t)N * 96), alloc((size_t)N * 96) };
  float* htbuf[2] = { alloc((size_t)N * 80), alloc((size_t)N * 80) };
  float* a0 = alloc((size_t)N * 112);
  float* a1 = alloc((size_t)N * 384);
  float* a2 = alloc((size_t)N * 400);
  // readout scratch aliased into a1/a2 (free after the message-passing layers)
  float* karr  = a1;
  float* varr  = a1 + (size_t)N * 64;
  float* qarr  = a1 + (size_t)N * 128;
  float* logit = qarr + (size_t)G * 64;
  float* zr    = a2;
  float* hid1  = zr + (size_t)G * 176;
  float* gden  = hid1 + (size_t)G * 128;
  unsigned* gmax = (unsigned*)(gden + G);
  float* gc    = gden + 2 * (size_t)G;

  // packed bf16 weight region (32B aligned)
  __bf16* packbase = (__bf16*)(ws + off);
  size_t poff = 0;
  auto packw = [&](const float* W, int K, int NCc) {
    int numKt = (K + 31) / 32, numNt = NCc / 16;
    size_t total = (size_t)numKt * numNt * 32;   // lane-fragments
    __bf16* dst = packbase + poff;
    poff += total * 16;
    pack_w_kernel<<<dim3((unsigned)((total + 63) / 64)), dim3(64), 0, stream>>>(W, dst, K, NCc);
    return dst;
  };

  const dim3 B256(256), B128(128);

  // pack all weights (constant across the launch)
  __bf16 *pWs[4], *pWss[4], *pWv[4], *pWvv[4], *pWt[4], *pWtt[4];
  for (int l = 0; l < 4; ++l) {
    pWs[l]  = packw(Ws  + (size_t)l * 7168, 112, 64);
    pWss[l] = packw(Wss + (size_t)l * 4096, 64, 64);
    pWv[l]  = packw(Wv  + (size_t)l * 4096, 128, 32);
    pWvv[l] = packw(Wvv + (size_t)l * 1024, 32, 32);
    pWt[l]  = packw(Wt  + (size_t)l * 1280, 80, 16);
    pWtt[l] = packw(Wtt + (size_t)l * 256, 16, 16);
  }
  __bf16* pWq  = packw(Wq, 64, 64);
  __bf16* pWk  = packw(Wk, 64, 64);
  __bf16* pWvp = packw(Wvp, 64, 64);
  __bf16* pWr1 = packw(Wr1, 176, 128);
  __bf16* pWr2 = packw(Wr2, 128, 128);

  edge_geom_kernel<<<dim3((E + 255) / 256), B256, 0, stream>>>(pos, ei, Ybuf, rbuf, E);
  hs_init_kernel<<<dim3((N * 64 + 255) / 256), B256, 0, stream>>>(z, embed, hsbuf[0], N);
  fill_f32<<<dim3(1024), B256, 0, stream>>>(hvbuf[0], 0.f, (size_t)N * 96);
  fill_f32<<<dim3(1024), B256, 0, stream>>>(htbuf[0], 0.f, (size_t)N * 80);

  int cur = 0;
  for (int l = 0; l < 4; ++l) {
    fill_f32<<<dim3(2048), B256, 0, stream>>>(a0, 0.f, (size_t)N * 112);
    fill_f32<<<dim3(2048), B256, 0, stream>>>(a1, 0.f, (size_t)N * 384);
    fill_f32<<<dim3(2048), B256, 0, stream>>>(a2, 0.f, (size_t)N * 400);
    message_kernel<<<dim3((E + 7) / 8), B256, 0, stream>>>(
        ei, Ybuf, rbuf, hsbuf[cur], hvbuf[cur], htbuf[cur],
        rW1 + (size_t)l * 320, rb1 + (size_t)l * 32, rW2 + (size_t)l * 10240,
        a0, a1, a2, E);
    const int nxt = cur ^ 1;
    {
      int tiles = (N + 15) / 16;
      gemm16_kernel<112, 64, 64, true, true, false, false>
          <<<dim3((tiles + 3) / 4), B128, 0, stream>>>(
          a0, 112, pWs[l], hsbuf[cur], 64, pWss[l],
          nullptr, nullptr, hsbuf[nxt], 64, N);
    }
    {
      int M = N * 3, tiles = (M + 15) / 16;
      gemm16_kernel<128, 32, 32, false, true, false, false>
          <<<dim3((tiles + 3) / 4), B128, 0, stream>>>(
          a1, 128, pWv[l], hvbuf[cur], 32, pWvv[l],
          nullptr, nullptr, hvbuf[nxt], 32, M);
    }
    {
      int M = N * 5, tiles = (M + 15) / 16;
      gemm16_kernel<80, 16, 16, false, true, false, false>
          <<<dim3((tiles + 3) / 4), B128, 0, stream>>>(
          a2, 80, pWt[l], htbuf[cur], 16, pWtt[l],
          nullptr, nullptr, htbuf[nxt], 16, M);
    }
    cur = nxt;
  }

  // attention readout
  {
    int tiles = (N + 15) / 16;
    gemm16_kernel<64, 0, 64, false, false, false, false>
        <<<dim3((tiles + 3) / 4), B128, 0, stream>>>(
        hsbuf[cur], 64, pWk, nullptr, 0, nullptr, nullptr, nullptr, karr, 64, N);
    gemm16_kernel<64, 0, 64, false, false, false, false>
        <<<dim3((tiles + 3) / 4), B128, 0, stream>>>(
        hsbuf[cur], 64, pWvp, nullptr, 0, nullptr, nullptr, nullptr, varr, 64, N);
  }
  {
    int tiles = (G + 15) / 16;
    gemm16_kernel<64, 0, 64, false, false, false, true>
        <<<dim3((tiles + 3) / 4), B128, 0, stream>>>(
        hsbuf[cur], 64, pWq, nullptr, 0, nullptr, nullptr, aidx, qarr, 64, G);
  }
  fill_u32<<<dim3(64), B256, 0, stream>>>(gmax, h_f2ord(-3.0e38f), (size_t)G);
  fill_f32<<<dim3(64), B256, 0, stream>>>(gden, 0.f, (size_t)G);
  fill_f32<<<dim3(128), B256, 0, stream>>>(gc, 0.f, (size_t)G * 64);
  logit_kernel<<<dim3((N + 255) / 256), B256, 0, stream>>>(batch, qarr, karr, logit, gmax, N);
  expsum_kernel<<<dim3((N + 255) / 256), B256, 0, stream>>>(batch, logit, gmax, varr, gden, gc, N);
  zr_kernel<<<dim3((G * 176 + 255) / 256), B256, 0, stream>>>(
      aidx, hsbuf[cur], hvbuf[cur], htbuf[cur], gc, gden, zr, G);
  {
    int tiles = (G + 15) / 16;
    gemm16_kernel<176, 0, 128, true, false, true, false>
        <<<dim3((tiles + 3) / 4), B128, 0, stream>>>(
        zr, 176, pWr1, nullptr, 0, nullptr, br1, nullptr, hid1, 128, G);
    gemm16_kernel<128, 0, 128, false, false, true, false>
        <<<dim3((tiles + 3) / 4), B128, 0, stream>>>(
        hid1, 128, pWr2, nullptr, 0, nullptr, br2, nullptr, (float*)d_out, 128, G);
  }
}
